// MultiheadSelfAttention_91311004713017
// MI455X (gfx1250) — compile-verified, matrix operand order checked by reading, not run on hardware
//
#include <hip/hip_runtime.h>
#include <hip/hip_bf16.h>

// B=2, L=2048, D=1024, H=16, DH=64. Causal multi-head self-attention.
// All GEMMs via v_wmma_f32_16x16x32_f16 (f16 operands, f32 accum).
// LDS tiles stored in *fragment order* (K bit3<->bit4 swapped, B operand
// operand-major [n][perm(k)]) so every WMMA operand is two ds_load_b128.
// f16 tiles are staged with GLOBAL_LOAD_ASYNC_TO_LDS_B32 (ASYNCcnt path).

#define BB 2
#define LL 2048
#define DD 1024
#define HH 16
#define DH 64

typedef __attribute__((ext_vector_type(16))) _Float16 v16h;
typedef __attribute__((ext_vector_type(8)))  _Float16 v8h;
typedef __attribute__((ext_vector_type(8)))  float    v8f;

__device__ __forceinline__ v8f wmma_f16(v16h a, v16h b, v8f c) {
    // (neg_a, A, neg_b, B, c_mod, C, reuse_a, reuse_b)
    return __builtin_amdgcn_wmma_f32_16x16x32_f16(false, a, false, b, (short)0, c, false, false);
}

// K-index permutation for the f16 fragment layout: swap bits 3 and 4.
__device__ __forceinline__ int permk(int k) {
    return (k & ~24) | ((k & 8) << 1) | ((k & 16) >> 1);
}
// For a 64-wide K dim (two 32-chunks): permute within each chunk.
__device__ __forceinline__ int permk64(int k) {
    return (k & 32) | permk(k & 31);
}

__device__ __forceinline__ unsigned pack2h(float a, float b) {
    union { _Float16 h[2]; unsigned u; } t;
    t.h[0] = (_Float16)a; t.h[1] = (_Float16)b;
    return t.u;
}

// Byte offset of a __shared__ pointer within the wave's LDS allocation.
__device__ __forceinline__ unsigned lds_off(void* p) {
    return (unsigned)(unsigned long long)(__attribute__((address_space(3))) void*)p;
}

// Per-lane async global->LDS DMA (GLOBAL_LOAD_ASYNC_TO_LDS_B32, ASYNCcnt).
__device__ __forceinline__ void async_ld_b32(unsigned lds, const void* gaddr) {
    asm volatile("global_load_async_to_lds_b32 %0, %1, off"
                 :: "v"(lds), "v"((unsigned long long)gaddr) : "memory");
}
__device__ __forceinline__ void wait_async0() {
    asm volatile("s_wait_asynccnt 0x0" ::: "memory");
}

// Load a 16x32 fragment: lane (grp,row) reads 16 contiguous permuted halfs
// = two aligned ds_load_b128.
__device__ __forceinline__ v16h load_frag(const _Float16* lds, int row, int stride,
                                          int kbase, int grp) {
    const _Float16* p = lds + row * stride + kbase + grp * 16;
    union { v16h v; v8h h[2]; } f;
    f.h[0] = *(const v8h*)(p);
    f.h[1] = *(const v8h*)(p + 8);
    return f.v;
}

// ---------------------------------------------------------------------------
// Kernel 1: fused QKV projection.  out[m,o] = sum_k x[m,k] * W[o,k]
// Writes head-split [B,H,L,DH] f16; Q pre-scaled by 1/sqrt(DH).
// Block: 256 threads (8 waves). Block tile 128(M) x 128(N), K step 32.
// (fp32 sources need conversion -> VALU staging path.)
// ---------------------------------------------------------------------------
__global__ __launch_bounds__(256) void qkv_proj_kernel(
    const float* __restrict__ x,
    const float* __restrict__ wq, const float* __restrict__ wk, const float* __restrict__ wv,
    _Float16* __restrict__ Qh, _Float16* __restrict__ Kh, _Float16* __restrict__ Vh)
{
    __shared__ _Float16 xs[128][40];   // M x perm(K)
    __shared__ _Float16 wt[128][40];   // N x perm(K)  (operand-major)

    const int tid  = threadIdx.x;
    const int lane = tid & 31;
    const int w    = tid >> 5;
    const int grp  = lane >> 4, nn = lane & 15;
    const int cb   = blockIdx.x * 128;         // output-column (o) base
    const int rb   = blockIdx.y * 128;         // row (b*L+l) base
    const int which = blockIdx.z;              // 0:Q 1:K 2:V

    const float* wgt = (which == 0) ? wq : (which == 1) ? wk : wv;
    _Float16*    dst = (which == 0) ? Qh : (which == 1) ? Kh : Vh;
    const float  scale = (which == 0) ? 0.125f : 1.0f;   // 1/sqrt(64)

    v8f acc[8];
#pragma unroll
    for (int n = 0; n < 8; ++n) acc[n] = (v8f){0,0,0,0,0,0,0,0};

    for (int k0 = 0; k0 < DD; k0 += 32) {
#pragma unroll
        for (int i = 0; i < 8; ++i) {          // 256 thr * 8 = 2048 half-pairs
            int idx = tid + i * 256;
            int r = idx >> 4;                  // 0..127
            int c = (idx & 15) * 2;            // even K
            int pc = permk(c);
            float2 xv = *(const float2*)&x[(size_t)(rb + r) * DD + k0 + c];
            *(unsigned*)&xs[r][pc] = pack2h(xv.x, xv.y);
            float2 wv2 = *(const float2*)&wgt[(size_t)(cb + r) * DD + k0 + c];
            *(unsigned*)&wt[r][pc] = pack2h(wv2.x, wv2.y);
        }
        __syncthreads();

        v16h a = load_frag(&xs[w * 16][0], nn, 40, 0, grp);
#pragma unroll
        for (int n = 0; n < 8; ++n) {
            v16h b = load_frag(&wt[n * 16][0], nn, 40, 0, grp);
            acc[n] = wmma_f16(a, b, acc[n]);
        }
        __syncthreads();
    }

    // C layout: element e -> row = 8*grp + e, col = lane&15 (within 16x16 tile)
#pragma unroll
    for (int n = 0; n < 8; ++n) {
#pragma unroll
        for (int e = 0; e < 8; ++e) {
            int m = rb + w * 16 + grp * 8 + e;      // global row in [0, B*L)
            int o = cb + n * 16 + nn;               // global output col
            int b = m >> 11, l = m & (LL - 1);
            int h = o >> 6,  d = o & (DH - 1);
            dst[(((size_t)(b * HH + h)) * LL + l) * DH + d] = (_Float16)(acc[n][e] * scale);
        }
    }
}

// ---------------------------------------------------------------------------
// Kernel 2: causal flash attention.
// Block = (64-query tile, head, batch), 128 threads = 4 waves, 16 q-rows/wave.
// Q/K tiles staged via async global->LDS (permuted scatter, no conversion);
// S = Q*K^T via WMMA, online softmax fp32, P re-laid via LDS, O += P*V WMMA.
// ---------------------------------------------------------------------------
__global__ __launch_bounds__(128) void flash_attn_kernel(
    const _Float16* __restrict__ Q, const _Float16* __restrict__ K,
    const _Float16* __restrict__ V, _Float16* __restrict__ AO)
{
    __shared__ _Float16 qs[64][72];      // q-row x perm64(d)
    __shared__ _Float16 kts[32][72];     // key  x perm64(d)   (B operand of S)
    __shared__ _Float16 vts[64][40];     // d    x perm(key)   (B operand of PV)
    __shared__ _Float16 ps[4][16][40];   // per-wave P: q-row x perm(key)

    const int tid  = threadIdx.x;
    const int lane = tid & 31;
    const int w    = tid >> 5;
    const int grp  = lane >> 4, nn = lane & 15;
    const int qt   = blockIdx.x;         // query tile (64 rows)
    const int h    = blockIdx.y;
    const int b    = blockIdx.z;

    const size_t headoff = ((size_t)(b * HH + h)) * LL * DH;
    const _Float16* Qp = Q + headoff + (size_t)qt * 64 * DH;
    const _Float16* Kp = K + headoff;
    const _Float16* Vp = V + headoff;

#pragma unroll
    for (int i = 0; i < 16; ++i) {       // 64x64 Q tile = 2048 half-pairs, async DMA
        int idx = tid + i * 128;
        int r = idx >> 5, c = (idx & 31) * 2;
        async_ld_b32(lds_off(&qs[r][permk64(c)]), &Qp[r * DH + c]);
    }
    wait_async0();
    __syncthreads();

    v16h aq0 = load_frag(&qs[w * 16][0], nn, 72,  0, grp);   // d 0..31
    v16h aq1 = load_frag(&qs[w * 16][0], nn, 72, 32, grp);   // d 32..63

    v8f oacc[4];
#pragma unroll
    for (int n = 0; n < 4; ++n) oacc[n] = (v8f){0,0,0,0,0,0,0,0};
    float mrow[8], lrow[8];
#pragma unroll
    for (int e = 0; e < 8; ++e) { mrow[e] = -1e30f; lrow[e] = 0.0f; }

    const int qrow_base = qt * 64 + w * 16 + grp * 8;   // + e

    const int nkb = 2 * qt + 2;          // causal: keys up to qt*64+63
    for (int kb = 0; kb < nkb; ++kb) {
        const int k0 = kb * 32;
        __syncthreads();                 // previous iteration fully consumed
#pragma unroll
        for (int i = 0; i < 8; ++i) {    // K tile: 32x64 = 1024 pairs, async DMA
            int idx = tid + i * 128;
            int kk = idx >> 5, c = (idx & 31) * 2;
            async_ld_b32(lds_off(&kts[kk][permk64(c)]),
                         &Kp[(size_t)(k0 + kk) * DH + c]);
        }
#pragma unroll
        for (int i = 0; i < 8; ++i) {    // V tile: key-pairs -> needs repack (VALU)
            int idx = tid + i * 128;
            int d = idx & 63, kk = (idx >> 6) * 2;
            _Float16 a0 = Vp[(size_t)(k0 + kk) * DH + d];
            _Float16 a1 = Vp[(size_t)(k0 + kk + 1) * DH + d];
            union { _Float16 h[2]; unsigned u; } t;
            t.h[0] = a0; t.h[1] = a1;
            *(unsigned*)&vts[d][permk(kk)] = t.u;
        }
        wait_async0();
        __syncthreads();

        // S tiles: 16 q-rows x 32 keys, f32 accum
        v8f s0 = (v8f){0,0,0,0,0,0,0,0};
        v8f s1 = (v8f){0,0,0,0,0,0,0,0};
        s0 = wmma_f16(aq0, load_frag(&kts[0][0],  nn, 72,  0, grp), s0);
        s0 = wmma_f16(aq1, load_frag(&kts[0][0],  nn, 72, 32, grp), s0);
        s1 = wmma_f16(aq0, load_frag(&kts[16][0], nn, 72,  0, grp), s1);
        s1 = wmma_f16(aq1, load_frag(&kts[16][0], nn, 72, 32, grp), s1);

        // causal mask + online softmax (row r lives across 16 lanes of a half-wave)
#pragma unroll
        for (int e = 0; e < 8; ++e) {
            const int qr  = qrow_base + e;
            const int kc0 = k0 + nn;
            const int kc1 = k0 + 16 + nn;
            float v0 = (kc0 <= qr) ? s0[e] : -1e30f;
            float v1 = (kc1 <= qr) ? s1[e] : -1e30f;
            float mx = fmaxf(v0, v1);
            mx = fmaxf(mx, __shfl_xor(mx, 1));
            mx = fmaxf(mx, __shfl_xor(mx, 2));
            mx = fmaxf(mx, __shfl_xor(mx, 4));
            mx = fmaxf(mx, __shfl_xor(mx, 8));
            float mnew  = fmaxf(mrow[e], mx);
            float alpha = __expf(mrow[e] - mnew);
            float p0 = __expf(v0 - mnew);
            float p1 = __expf(v1 - mnew);
            float rs = p0 + p1;
            rs += __shfl_xor(rs, 1);
            rs += __shfl_xor(rs, 2);
            rs += __shfl_xor(rs, 4);
            rs += __shfl_xor(rs, 8);
            lrow[e] = lrow[e] * alpha + rs;
            mrow[e] = mnew;
#pragma unroll
            for (int n = 0; n < 4; ++n) oacc[n][e] *= alpha;
            ps[w][grp * 8 + e][permk(nn)]      = (_Float16)p0;  // C-layout -> permuted LDS
            ps[w][grp * 8 + e][permk(16 + nn)] = (_Float16)p1;
        }
        __syncthreads();

        v16h pa = load_frag(&ps[w][0][0], nn, 40, 0, grp);      // A-frag of P
#pragma unroll
        for (int n = 0; n < 4; ++n) {
            v16h bv = load_frag(&vts[n * 16][0], nn, 40, 0, grp);
            oacc[n] = wmma_f16(pa, bv, oacc[n]);
        }
    }

    // epilogue: O /= l, store f16 in [B, L, H, DH] (== row-major [B*L, D])
#pragma unroll
    for (int e = 0; e < 8; ++e) {
        float inv = 1.0f / fmaxf(lrow[e], 1e-20f);
        int ql = qt * 64 + w * 16 + grp * 8 + e;
#pragma unroll
        for (int n = 0; n < 4; ++n) {
            int d = n * 16 + nn;
            AO[(((size_t)b * LL + ql) * HH + h) * DH + d] = (_Float16)(oacc[n][e] * inv);
        }
    }
}

// ---------------------------------------------------------------------------
// Kernel 3: output projection. out[m,o] = sum_k AO[m,k] * Wo[o,k], fp32 out.
// AO tile (f16) staged via async global->LDS DMA; Wo converted on VALU path.
// ---------------------------------------------------------------------------
__global__ __launch_bounds__(256) void out_proj_kernel(
    const _Float16* __restrict__ ao, const float* __restrict__ wo,
    float* __restrict__ out)
{
    __shared__ _Float16 xs[128][40];   // M x perm(K)
    __shared__ _Float16 wt[128][40];   // N x perm(K)

    const int tid  = threadIdx.x;
    const int lane = tid & 31;
    const int w    = tid >> 5;
    const int grp  = lane >> 4, nn = lane & 15;
    const int cb   = blockIdx.x * 128;
    const int rb   = blockIdx.y * 128;

    v8f acc[8];
#pragma unroll
    for (int n = 0; n < 8; ++n) acc[n] = (v8f){0,0,0,0,0,0,0,0};

    for (int k0 = 0; k0 < DD; k0 += 32) {
#pragma unroll
        for (int i = 0; i < 8; ++i) {
            int idx = tid + i * 256;
            int r = idx >> 4;
            int c = (idx & 15) * 2;
            int pc = permk(c);
            async_ld_b32(lds_off(&xs[r][pc]), &ao[(size_t)(rb + r) * DD + k0 + c]);
            float2 wv2 = *(const float2*)&wo[(size_t)(cb + r) * DD + k0 + c];
            *(unsigned*)&wt[r][pc] = pack2h(wv2.x, wv2.y);
        }
        wait_async0();
        __syncthreads();

        v16h a = load_frag(&xs[w * 16][0], nn, 40, 0, grp);
#pragma unroll
        for (int n = 0; n < 8; ++n) {
            v16h b = load_frag(&wt[n * 16][0], nn, 40, 0, grp);
            acc[n] = wmma_f16(a, b, acc[n]);
        }
        __syncthreads();
    }

#pragma unroll
    for (int n = 0; n < 8; ++n) {
#pragma unroll
        for (int e = 0; e < 8; ++e) {
            int m = rb + w * 16 + grp * 8 + e;
            int o = cb + n * 16 + nn;
            out[(size_t)m * DD + o] = acc[n][e];
        }
    }
}

// ---------------------------------------------------------------------------
extern "C" void kernel_launch(void* const* d_in, const int* in_sizes, int n_in,
                              void* d_out, int out_size, void* d_ws, size_t ws_size,
                              hipStream_t stream) {
    const float* x  = (const float*)d_in[0];
    // d_in[1] is the causal tril mask; causality is applied analytically.
    const float* wq = (const float*)d_in[2];
    const float* wk = (const float*)d_in[3];
    const float* wv = (const float*)d_in[4];
    const float* wo = (const float*)d_in[5];

    const size_t nElem = (size_t)BB * HH * LL * DH;   // 4M elements each
    _Float16* Qh = (_Float16*)d_ws;
    _Float16* Kh = Qh + nElem;
    _Float16* Vh = Kh + nElem;
    _Float16* AO = Vh + nElem;                         // total 32 MB f16

    qkv_proj_kernel<<<dim3(DD / 128, (BB * LL) / 128, 3), 256, 0, stream>>>(
        x, wq, wk, wv, Qh, Kh, Vh);

    flash_attn_kernel<<<dim3(LL / 64, HH, BB), 128, 0, stream>>>(Qh, Kh, Vh, AO);

    out_proj_kernel<<<dim3(DD / 128, (BB * LL) / 128, 1), 256, 0, stream>>>(
        AO, wo, (float*)d_out);
}
